// ConvolutionalNMPBlock_28312424415492
// MI455X (gfx1250) — matrix operand order repaired and checked by reading, not compiled, not run on hardware
//
#include <hip/hip_runtime.h>
#include <hip/hip_bf16.h>

// ---------------------------------------------------------------------------
// ConvolutionalNMPBlock on MI455X (gfx1250, wave32, WMMA bf16 -> f32 acc)
// b=16, n=2048, d=256.  Heavy GEMMs via v_wmma_f32_16x16x32_bf16.
// conv2 activation tiles staged to LDS with the Tensor Data Mover (TDM).
// ---------------------------------------------------------------------------

#define BSZ 16
#define NSEQ 2048
#define DIM 256
#define NR (BSZ * NSEQ)      // 32768 rows
#define NPAD (NSEQ + 16)     // h1 padded with 8 zero rows each side per batch
#define EPS 1e-5f

typedef __attribute__((ext_vector_type(16))) __bf16 v16bf;
typedef __attribute__((ext_vector_type(8)))  float  v8f;
typedef __attribute__((ext_vector_type(4)))  unsigned int v4u;
typedef __attribute__((ext_vector_type(8)))  int    v8i;
typedef __attribute__((ext_vector_type(4)))  int    v4i;

union Frag16 { v16bf v; uint4 q[2]; };

// round-to-nearest-even f32 -> bf16 (stored as u16)
__device__ __forceinline__ unsigned short f2bf(float f) {
    union { float f; unsigned u; } a; a.f = f;
    unsigned r = a.u + 0x7FFFu + ((a.u >> 16) & 1u);
    return (unsigned short)(r >> 16);
}

// A fragment: 16x32 bf16, row-major source, leading dim lda (elements).
// Lane l (l<16): row, K chunks [0..7],[16..23]; lane l+16: K [8..15],[24..31].
__device__ __forceinline__ v16bf frag_a(const unsigned short* A, int lda,
                                        int row, int kbase, int lane) {
    const unsigned short* p = A + (size_t)row * lda + kbase + ((lane >= 16) ? 8 : 0);
    Frag16 f;
    f.q[0] = *(const uint4*)(p);
    f.q[1] = *(const uint4*)(p + 16);
    return f.v;
}

// B fragment from W stored row-major as (N, K) (i.e. out = A @ W^T).
// Lane l: column = l&15; lanes 0..15 hold K [0..7],[16..23]; lanes 16..31 hold K [8..15],[24..31].
__device__ __forceinline__ v16bf frag_bT(const unsigned short* W, int ldw,
                                         int ncol0, int kbase, int lane) {
    const unsigned short* p = W + (size_t)(ncol0 + (lane & 15)) * ldw
                              + kbase + ((lane >= 16) ? 8 : 0);
    Frag16 f;
    f.q[0] = *(const uint4*)(p);
    f.q[1] = *(const uint4*)(p + 16);
    return f.v;
}

__device__ __forceinline__ v8f wmma_bf16(v16bf a, v16bf b, v8f c) {
    return __builtin_amdgcn_wmma_f32_16x16x32_bf16(false, a, false, b,
                                                   (short)0, c, false, false);
}

// ---------------------------------------------------------------------------
// Packing kernels
// ---------------------------------------------------------------------------
__global__ void k_f32_to_bf16(const float* __restrict__ s,
                              unsigned short* __restrict__ d, int n) {
    for (int i = blockIdx.x * blockDim.x + threadIdx.x; i < n;
         i += gridDim.x * blockDim.x)
        d[i] = f2bf(s[i]);
}

// conv2_w (cout, cin, 17) -> bf16 [k][cout][cin]
__global__ void k_pack_w2(const float* __restrict__ w2,
                          unsigned short* __restrict__ d) {
    int i = blockIdx.x * blockDim.x + threadIdx.x;
    if (i >= 17 * DIM * DIM) return;
    int k = i >> 16, cout = (i >> 8) & 255, cin = i & 255;
    d[i] = f2bf(w2[((size_t)cout * DIM + cin) * 17 + k]);
}

// zero the 8-row halo at both ends of each batch in padded h1
__global__ void k_zero_pad(unsigned short* __restrict__ h1p) {
    int i = blockIdx.x * blockDim.x + threadIdx.x;   // BSZ*16*DIM elements
    if (i >= BSZ * 16 * DIM) return;
    int b = i / (16 * DIM), rr = (i / DIM) & 15, c = i & 255;
    int row = (rr < 8) ? rr : (NPAD - 16 + rr);
    h1p[((size_t)b * NPAD + row) * DIM + c] = 0;
}

// s = x @ w_se^T + b_se ;  store float4 {s0,s1,s2, |s|^2}
__global__ void k_se(const float* __restrict__ x, const float* __restrict__ w_se,
                     const float* __restrict__ b_se, float4* __restrict__ s4) {
    int r = blockIdx.x * blockDim.x + threadIdx.x;
    if (r >= NR) return;
    const float* xr = x + (size_t)r * DIM;
    float s0 = b_se[0], s1 = b_se[1], s2 = b_se[2];
    for (int c = 0; c < DIM; ++c) {
        float xv = xr[c];
        s0 += w_se[c] * xv;
        s1 += w_se[DIM + c] * xv;
        s2 += w_se[2 * DIM + c] * xv;
    }
    s4[r] = make_float4(s0, s1, s2, s0 * s0 + s1 * s1 + s2 * s2);
}

// ---------------------------------------------------------------------------
// conv1 (1x1) + BN1 + ReLU : h1 = relu(bn1(x @ w1^T))  -> padded bf16 layout
// block = 128 (4 waves); grid = (NR/16, DIM/64)
// ---------------------------------------------------------------------------
__global__ void k_conv1(const unsigned short* __restrict__ xb,
                        const unsigned short* __restrict__ w1b,
                        const float* __restrict__ g, const float* __restrict__ bb,
                        const float* __restrict__ m, const float* __restrict__ v,
                        unsigned short* __restrict__ h1p) {
    int lane = threadIdx.x & 31, wave = threadIdx.x >> 5;
    int row0 = blockIdx.x * 16;
    int col0 = blockIdx.y * 64 + wave * 16;
    v8f acc = {};
    int rowl = row0 + (lane & 15);
    for (int k = 0; k < DIM; k += 32) {
        v16bf a = frag_a(xb, DIM, rowl, k, lane);
        v16bf b = frag_bT(w1b, DIM, col0, k, lane);
        acc = wmma_bf16(a, b, acc);
    }
    int col = col0 + (lane & 15);
    float scale = g[col] * rsqrtf(v[col] + EPS);
    float shift = bb[col] - m[col] * scale;
    int r0 = row0 + ((lane >= 16) ? 8 : 0);
    for (int j = 0; j < 8; ++j) {
        int rg = r0 + j;
        int bi = rg >> 11, tt = rg & (NSEQ - 1);
        float val = fmaxf(acc[j] * scale + shift, 0.f);
        h1p[((size_t)bi * NPAD + 8 + tt) * DIM + col] = f2bf(val);
    }
}

// ---------------------------------------------------------------------------
// conv2 (k=17, pad 8) + BN2 + residual + ReLU -> x_in[:, 0:256]  (bf16)
// TDM loads the 32x256 bf16 activation tile (rows t0-8 .. t0+23, in padded
// coords t0 .. t0+31) into LDS once per block; 17 row-shifted GEMMs read it.
// ---------------------------------------------------------------------------
__global__ void k_conv2(const unsigned short* __restrict__ h1p,
                        const unsigned short* __restrict__ w2b,
                        const float* __restrict__ g, const float* __restrict__ bb,
                        const float* __restrict__ m, const float* __restrict__ v,
                        const float* __restrict__ x,
                        unsigned short* __restrict__ x_in) {
    __shared__ alignas(16) unsigned short Atile[32 * DIM];  // sole LDS object -> offset 0
    int lane = threadIdx.x & 31, wave = threadIdx.x >> 5;
    int rt = blockIdx.x;
    int bidx = rt / (NSEQ / 16);
    int t0 = (rt % (NSEQ / 16)) * 16;
    int col0 = blockIdx.y * 64 + wave * 16;

    if (wave == 0) {
        // ---- Tensor DMA descriptor (D#), ISA ch.8: 2D tile 256 x 32 ----
        unsigned long long ga =
            (unsigned long long)(const void*)(h1p + ((size_t)bidx * NPAD + t0) * DIM);
        v4u g0;
        g0[0] = 1u;                                   // count=1, user mode
        g0[1] = 0u;                                   // lds_addr = 0 (Atile)
        g0[2] = (unsigned)ga;                         // global_addr[31:0]
        g0[3] = (unsigned)((ga >> 32) & 0x1FFFFFFu)   // global_addr[56:32]
                | (2u << 30);                         // type = 2 ("image")
        v8i g1;
        g1[0] = (int)(1u << 16);                      // wg_mask=0, data_size=1 (2B)
        g1[1] = (int)(((unsigned)DIM & 0xFFFFu) << 16);   // tensor_dim0[15:0]=256
        g1[2] = (int)(((unsigned)NPAD & 0xFFFFu) << 16);  // dim0[31:16]=0 | dim1[15:0]=2064
        g1[3] = (int)(((unsigned)DIM) << 16);         // dim1[31:16]=0 | tile_dim0=256
        g1[4] = 32;                                   // tile_dim1=32, tile_dim2=0
        g1[5] = DIM;                                  // tensor_dim0_stride[31:0]=256
        g1[6] = 0;                                    // stride0 hi | stride1 lo
        g1[7] = 0;                                    // stride1 hi
        v4i g2; g2[0] = 1; g2[1] = 1; g2[2] = 0; g2[3] = 0;   // dim2=dim3=1, tile_dim3=0
        v4i g3; g3[0] = 0; g3[1] = 0x00010000; g3[2] = 0; g3[3] = 0; // dim4=1, tile_dim4=0
        v8i gx = {};                                  // extra group (6-arg form), zero
        __builtin_amdgcn_tensor_load_to_lds(g0, g1, g2, g3, gx, 0);
        __builtin_amdgcn_s_wait_tensorcnt(0);
    }
    __syncthreads();

    v8f acc = {};
    for (int k = 0; k < 17; ++k) {
        const unsigned short* wk = w2b + (size_t)k * DIM * DIM;
        if (k < 16)   // pull next tap's weight panel toward the caches
            __builtin_prefetch(w2b + (size_t)(k + 1) * DIM * DIM + (size_t)col0 * DIM, 0, 3);
        int trow = (lane & 15) + k;                   // row in Atile (always in-bounds)
        for (int kk = 0; kk < DIM; kk += 32) {
            v16bf a = frag_a(Atile, DIM, trow, kk, lane);
            v16bf b = frag_bT(wk, DIM, col0, kk, lane);
            acc = wmma_bf16(a, b, acc);
        }
    }

    int col = col0 + (lane & 15);
    float scale = g[col] * rsqrtf(v[col] + EPS);
    float shift = bb[col] - m[col] * scale;
    int r0 = bidx * NSEQ + t0 + ((lane >= 16) ? 8 : 0);
    for (int j = 0; j < 8; ++j) {
        size_t r = (size_t)(r0 + j);
        float val = fmaxf(acc[j] * scale + shift + x[r * DIM + col], 0.f);
        x_in[r * 2 * DIM + col] = f2bf(val);
    }
}

// ---------------------------------------------------------------------------
// msg = relu(x @ w_msg^T + b_msg)  (bf16)
// ---------------------------------------------------------------------------
__global__ void k_msg(const unsigned short* __restrict__ xb,
                      const unsigned short* __restrict__ wmb,
                      const float* __restrict__ bias,
                      unsigned short* __restrict__ msgb) {
    int lane = threadIdx.x & 31, wave = threadIdx.x >> 5;
    int row0 = blockIdx.x * 16;
    int col0 = blockIdx.y * 64 + wave * 16;
    v8f acc = {};
    int rowl = row0 + (lane & 15);
    for (int k = 0; k < DIM; k += 32) {
        v16bf a = frag_a(xb, DIM, rowl, k, lane);
        v16bf b = frag_bT(wmb, DIM, col0, k, lane);
        acc = wmma_bf16(a, b, acc);
    }
    int col = col0 + (lane & 15);
    float bi = bias[col];
    int r0 = row0 + ((lane >= 16) ? 8 : 0);
    for (int j = 0; j < 8; ++j) {
        float val = fmaxf(acc[j] + bi, 0.f);
        msgb[(size_t)(r0 + j) * DIM + col] = f2bf(val);
    }
}

// ---------------------------------------------------------------------------
// NMP: x_nmp = (exp(-dist) * mask) @ msg  -> x_in[:, 256:512]
// Flash-style: A tile (16x32) built in LDS, msg tile transposed in LDS,
// 8 waves each own 32 output columns. A never touches HBM.
// block = 256 (8 waves); grid = BSZ * NSEQ/16
// ---------------------------------------------------------------------------
__global__ void k_nmp(const float4* __restrict__ s4,
                      const float* __restrict__ mask,
                      const unsigned short* __restrict__ msgb,
                      unsigned short* __restrict__ x_in) {
    __shared__ alignas(16) unsigned short As[16 * 32];
    __shared__ alignas(16) unsigned short Bt[256 * 32];
    __shared__ float4 sT[16];
    int tid = threadIdx.x, lane = tid & 31, wave = tid >> 5;
    int bidx = blockIdx.x / (NSEQ / 16);
    int t0 = (blockIdx.x % (NSEQ / 16)) * 16;
    if (tid < 16) sT[tid] = s4[bidx * NSEQ + t0 + tid];
    __syncthreads();

    v8f acc0 = {}, acc1 = {};
    const float* mbase = mask + (size_t)bidx * NSEQ * NSEQ;

    for (int m0 = 0; m0 < NSEQ; m0 += 32) {
        // --- build A tile: exp(-dist) * mask  (512 elems / 256 threads) ---
        #pragma unroll
        for (int ee = 0; ee < 2; ++ee) {
            int e = tid + ee * 256;
            int i = e >> 5, kk = e & 31;
            float4 sm = s4[bidx * NSEQ + m0 + kk];
            float4 st = sT[i];
            float dist = st.w + sm.w - 2.f * (st.x * sm.x + st.y * sm.y + st.z * sm.z);
            float aval = __expf(-dist) * mbase[(size_t)(t0 + i) * NSEQ + m0 + kk];
            As[i * 32 + kk] = f2bf(aval);
        }
        // --- stage msg tile transposed: Bt[col][m]  (32 rows x 256 cols) ---
        {
            int mr = tid >> 3, cg = tid & 7;   // row in tile, 32-col group
            const unsigned short* p = msgb + (size_t)(bidx * NSEQ + m0 + mr) * DIM + cg * 32;
            unsigned short tmp[32];
            *(uint4*)(tmp)      = *(const uint4*)(p);
            *(uint4*)(tmp + 8)  = *(const uint4*)(p + 8);
            *(uint4*)(tmp + 16) = *(const uint4*)(p + 16);
            *(uint4*)(tmp + 24) = *(const uint4*)(p + 24);
            #pragma unroll
            for (int c = 0; c < 32; ++c) Bt[(cg * 32 + c) * 32 + mr] = tmp[c];
        }
        __syncthreads();
        v16bf a  = frag_a(As, 32, lane & 15, 0, lane);
        v16bf b0 = frag_bT(Bt, 32, wave * 32, 0, lane);
        v16bf b1 = frag_bT(Bt, 32, wave * 32 + 16, 0, lane);
        acc0 = wmma_bf16(a, b0, acc0);
        acc1 = wmma_bf16(a, b1, acc1);
        __syncthreads();
    }

    int r0 = bidx * NSEQ + t0 + ((lane >= 16) ? 8 : 0);
    int c0 = wave * 32 + (lane & 15);
    for (int j = 0; j < 8; ++j) {
        size_t r = (size_t)(r0 + j);
        x_in[r * 2 * DIM + DIM + c0]      = f2bf(acc0[j]);
        x_in[r * 2 * DIM + DIM + c0 + 16] = f2bf(acc1[j]);
    }
}

// ---------------------------------------------------------------------------
// Fused GRU: gi = x_in @ w_ih^T, gh = x @ w_hh^T, gate math, f32 out.
// Each wave owns a 16x16 column tile of d; accumulates r/z/gi_n/gh_n (72 WMMA).
// ---------------------------------------------------------------------------
__global__ void k_gru(const unsigned short* __restrict__ x_in,
                      const unsigned short* __restrict__ xb,
                      const unsigned short* __restrict__ wih,
                      const unsigned short* __restrict__ whh,
                      const float* __restrict__ b_ih, const float* __restrict__ b_hh,
                      const float* __restrict__ x, float* __restrict__ out) {
    int lane = threadIdx.x & 31, wave = threadIdx.x >> 5;
    int row0 = blockIdx.x * 16;
    int col0 = blockIdx.y * 64 + wave * 16;
    v8f ar = {}, az = {}, ain = {}, ahn = {};
    int rowl = row0 + (lane & 15);
    for (int k = 0; k < 2 * DIM; k += 32) {       // gi over K=512
        v16bf a = frag_a(x_in, 2 * DIM, rowl, k, lane);
        ar  = wmma_bf16(a, frag_bT(wih, 2 * DIM, col0,           k, lane), ar);
        az  = wmma_bf16(a, frag_bT(wih, 2 * DIM, col0 + DIM,     k, lane), az);
        ain = wmma_bf16(a, frag_bT(wih, 2 * DIM, col0 + 2 * DIM, k, lane), ain);
    }
    for (int k = 0; k < DIM; k += 32) {           // gh over K=256
        v16bf a = frag_a(xb, DIM, rowl, k, lane);
        ar  = wmma_bf16(a, frag_bT(whh, DIM, col0,           k, lane), ar);
        az  = wmma_bf16(a, frag_bT(whh, DIM, col0 + DIM,     k, lane), az);
        ahn = wmma_bf16(a, frag_bT(whh, DIM, col0 + 2 * DIM, k, lane), ahn);
    }
    int col = col0 + (lane & 15);
    float br  = b_ih[col] + b_hh[col];
    float bz  = b_ih[DIM + col] + b_hh[DIM + col];
    float bin = b_ih[2 * DIM + col];
    float bhn = b_hh[2 * DIM + col];
    int r0 = row0 + ((lane >= 16) ? 8 : 0);
    for (int j = 0; j < 8; ++j) {
        size_t r = (size_t)(r0 + j);
        float xv = x[r * DIM + col];
        float rg = 1.f / (1.f + __expf(-(ar[j] + br)));
        float zg = 1.f / (1.f + __expf(-(az[j] + bz)));
        float nn = tanhf(ain[j] + bin + rg * (ahn[j] + bhn));
        out[r * DIM + col] = (1.f - zg) * nn + zg * xv;
    }
}

// ---------------------------------------------------------------------------
extern "C" void kernel_launch(void* const* d_in, const int* in_sizes, int n_in,
                              void* d_out, int out_size, void* d_ws, size_t ws_size,
                              hipStream_t stream) {
    const float* x       = (const float*)d_in[0];
    const float* mask    = (const float*)d_in[1];
    const float* w_se    = (const float*)d_in[2];
    const float* b_se    = (const float*)d_in[3];
    const float* conv1_w = (const float*)d_in[4];
    const float* bn1_g   = (const float*)d_in[5];
    const float* bn1_b   = (const float*)d_in[6];
    const float* bn1_m   = (const float*)d_in[7];
    const float* bn1_v   = (const float*)d_in[8];
    const float* conv2_w = (const float*)d_in[9];
    const float* bn2_g   = (const float*)d_in[10];
    const float* bn2_b   = (const float*)d_in[11];
    const float* bn2_m   = (const float*)d_in[12];
    const float* bn2_v   = (const float*)d_in[13];
    const float* w_msg   = (const float*)d_in[14];
    const float* b_msg   = (const float*)d_in[15];
    const float* w_ih    = (const float*)d_in[16];
    const float* b_ih    = (const float*)d_in[17];
    const float* w_hh    = (const float*)d_in[18];
    const float* b_hh    = (const float*)d_in[19];
    float* out = (float*)d_out;

    char* ws = (char*)d_ws;
    size_t off = 0;
    auto carve = [&](size_t bytes) {
        char* p = ws + off;
        off = (off + bytes + 255) & ~(size_t)255;
        return p;
    };
    unsigned short* xb    = (unsigned short*)carve((size_t)NR * DIM * 2);
    unsigned short* h1p   = (unsigned short*)carve((size_t)BSZ * NPAD * DIM * 2);
    unsigned short* x_in  = (unsigned short*)carve((size_t)NR * 2 * DIM * 2);
    unsigned short* msgb  = (unsigned short*)carve((size_t)NR * DIM * 2);
    float4*         s4    = (float4*)carve((size_t)NR * 16);
    unsigned short* w1b   = (unsigned short*)carve((size_t)DIM * DIM * 2);
    unsigned short* w2b   = (unsigned short*)carve((size_t)17 * DIM * DIM * 2);
    unsigned short* wmsgb = (unsigned short*)carve((size_t)DIM * DIM * 2);
    unsigned short* wihb  = (unsigned short*)carve((size_t)3 * DIM * 2 * DIM * 2);
    unsigned short* whhb  = (unsigned short*)carve((size_t)3 * DIM * DIM * 2);

    // ---- pack activations + weights to bf16, zero conv halo ----
    k_f32_to_bf16<<<4096, 256, 0, stream>>>(x, xb, NR * DIM);
    k_f32_to_bf16<<<256, 256, 0, stream>>>(conv1_w, w1b, DIM * DIM);
    k_pack_w2<<<(17 * DIM * DIM + 255) / 256, 256, 0, stream>>>(conv2_w, w2b);
    k_f32_to_bf16<<<256, 256, 0, stream>>>(w_msg, wmsgb, DIM * DIM);
    k_f32_to_bf16<<<1536, 256, 0, stream>>>(w_ih, wihb, 3 * DIM * 2 * DIM);
    k_f32_to_bf16<<<768, 256, 0, stream>>>(w_hh, whhb, 3 * DIM * DIM);
    k_zero_pad<<<(BSZ * 16 * DIM + 255) / 256, 256, 0, stream>>>(h1p);
    k_se<<<NR / 256, 256, 0, stream>>>(x, w_se, b_se, s4);

    dim3 ggrid(NR / 16, DIM / 64);   // (2048, 4), 4 waves/block
    k_conv1<<<ggrid, 128, 0, stream>>>(xb, w1b, bn1_g, bn1_b, bn1_m, bn1_v, h1p);
    k_conv2<<<ggrid, 128, 0, stream>>>(h1p, w2b, bn2_g, bn2_b, bn2_m, bn2_v, x, x_in);
    k_msg<<<ggrid, 128, 0, stream>>>(xb, wmsgb, b_msg, msgb);
    k_nmp<<<BSZ * (NSEQ / 16), 256, 0, stream>>>(s4, mask, msgb, x_in);
    k_gru<<<ggrid, 128, 0, stream>>>(x_in, xb, wihb, whhb, b_ih, b_hh, x, out);
}
